// SparseMoENetwork_59012850647400
// MI455X (gfx1250) — matrix operand order repaired and checked by reading, not compiled, not run on hardware
//
#include <hip/hip_runtime.h>
#include <hip/hip_bf16.h>

// Problem constants (from reference)
#define BB   4096
#define INF  512
#define TT   10
#define EE   16
#define SS   2
#define KK   4
#define WW   1024
#define HD   64
#define XROW (INF + TT)   // 522

typedef _Float16 half16 __attribute__((ext_vector_type(16)));
typedef _Float16 half8  __attribute__((ext_vector_type(8)));
typedef float    float8 __attribute__((ext_vector_type(8)));
typedef int      vec4i  __attribute__((vector_size(16)));   // matches async-LDS builtin param

#if defined(__has_builtin)
#if __has_builtin(__builtin_amdgcn_global_load_async_to_lds_b128)
#define HAVE_ASYNC_LDS 1
#endif
#endif

// ---------------- gating: logits, top-4 softmax -> dense gate weights ----------------
__global__ void gating_kernel(const float* __restrict__ x, const float* __restrict__ gk,
                              float* __restrict__ gateW, int* __restrict__ taskIdx) {
    int b = blockIdx.x * blockDim.x + threadIdx.x;
    if (b >= BB) return;
    const float* xr = x + (size_t)b * XROW;
    float xt[TT];
    int am = 0; float amv = xr[INF];
    #pragma unroll
    for (int t = 0; t < TT; ++t) {
        xt[t] = xr[INF + t];
        if (xt[t] > amv) { amv = xt[t]; am = t; }
    }
    taskIdx[b] = am;
    float lg[EE];
    #pragma unroll
    for (int e = 0; e < EE; ++e) {
        float s = 0.f;
        #pragma unroll
        for (int t = 0; t < TT; ++t) s += xt[t] * gk[t * EE + e];
        lg[e] = s;
    }
    int idx[KK]; float val[KK]; unsigned used = 0;
    #pragma unroll
    for (int k = 0; k < KK; ++k) {
        int bi = 0; float bv = -3.0e38f;
        #pragma unroll
        for (int e = 0; e < EE; ++e)
            if (!((used >> e) & 1u) && lg[e] > bv) { bv = lg[e]; bi = e; }
        used |= (1u << bi); idx[k] = bi; val[k] = bv;
    }
    float mx = val[0];
    float se = 0.f;
    #pragma unroll
    for (int k = 0; k < KK; ++k) { val[k] = __expf(val[k] - mx); se += val[k]; }
    float inv = 1.f / se;
    float w[EE];
    #pragma unroll
    for (int e = 0; e < EE; ++e) w[e] = 0.f;
    #pragma unroll
    for (int k = 0; k < KK; ++k) w[idx[k]] = val[k] * inv;
    #pragma unroll
    for (int e = 0; e < EE; ++e) gateW[(size_t)b * EE + e] = w[e];
}

// ---------------- feat f32 -> f16 ----------------
__global__ void cvt_feat_kernel(const float* __restrict__ x, _Float16* __restrict__ featH) {
    long i = (long)blockIdx.x * 256 + threadIdx.x;   // over B*INF
    int b = (int)(i >> 9);
    int c = (int)(i & 511);
    featH[i] = (_Float16)x[(size_t)b * XROW + c];
}

// ---------------- weight f32 [nE][R][C] -> f16 transposed [nE][C][R] ----------------
__global__ void tcvt_kernel(const float* __restrict__ src, _Float16* __restrict__ dst,
                            int R, int C) {
    long i  = (long)blockIdx.x * 256 + threadIdx.x;
    long rc = (long)R * C;
    long e  = i / rc;
    long rem = i - e * rc;
    int r = (int)(rem / C);
    int c = (int)(rem % C);
    dst[e * rc + (long)c * R + r] = (_Float16)src[i];
}

// ---------------- fused two-layer expert MLP (WMMA) ----------------
#define TM 64
#define NTHREADS 512            // 16 waves: 4 row-tiles x 4 col-quarters
#define IN_STRIDE  520          // 512 + pad (16B-aligned rows, avoids bank conflicts)
#define H1_STRIDE  1040         // 1024 + pad

__global__ __launch_bounds__(NTHREADS) void expert_kernel(
    const _Float16* __restrict__ featH,  // [B][512]
    const _Float16* __restrict__ r0T,    // [E][1024][512]
    const float*    __restrict__ rb0,    // [E][1024]
    const _Float16* __restrict__ r1T,    // [E][1024][1024]
    const float*    __restrict__ rb1,
    const _Float16* __restrict__ s0T,    // [S][1024][512]
    const float*    __restrict__ sb0,
    const _Float16* __restrict__ s1T,    // [S][1024][1024]
    const float*    __restrict__ sb1,
    const float*    __restrict__ gateW,  // [B][E]
    float*          __restrict__ acc)    // [B][1024]
{
    __shared__ _Float16 ldsIn[TM * IN_STRIDE];   // ~66.5 KB
    __shared__ _Float16 ldsH1[TM * H1_STRIDE];   // ~133 KB
    __shared__ float    ldsGw[TM];

    const int bBase = blockIdx.x * TM;
    const int e     = blockIdx.y;
    const bool isShared = (e >= EE);
    const int  se = e - EE;

    const _Float16* W0 = isShared ? s0T + (size_t)se * WW * INF : r0T + (size_t)e * WW * INF;
    const float*    B0 = isShared ? sb0 + (size_t)se * WW       : rb0 + (size_t)e * WW;
    const _Float16* W1 = isShared ? s1T + (size_t)se * WW * WW  : r1T + (size_t)e * WW * WW;
    const float*    B1 = isShared ? sb1 + (size_t)se * WW       : rb1 + (size_t)e * WW;

    const int tid = threadIdx.x;

    // ---- stage feat tile (f16) into LDS ----
#ifdef HAVE_ASYNC_LDS
    // async copy: 16B chunks; TM*512 halves = TM*64 chunks
    for (int i = tid; i < TM * 64; i += NTHREADS) {
        int row  = i >> 6;
        int colh = (i & 63) * 8;     // half index within the row
        __builtin_amdgcn_global_load_async_to_lds_b128(
            (vec4i*)(featH + (size_t)(bBase + row) * INF + colh),
            (vec4i*)&ldsIn[row * IN_STRIDE + colh],
            0, 0);
    }
    asm volatile("s_wait_asynccnt 0x0" ::: "memory");
#else
    for (int i = tid; i < TM * INF; i += NTHREADS) {
        int row = i >> 9, col = i & 511;
        ldsIn[row * IN_STRIDE + col] = featH[(size_t)(bBase + row) * INF + col];
    }
#endif
    if (tid < TM)
        ldsGw[tid] = isShared ? (1.0f / SS) : gateW[(size_t)(bBase + tid) * EE + e];
    __syncthreads();

    // 16 waves: wave = (row-tile mt in {0..3}) x (col-quarter nq in {0..3})
    const int wave = tid >> 5;
    const int lane = tid & 31;
    const int mt   = wave & 3;
    const int nq   = wave >> 2;
    const int rowBase = mt * 16;
    const int nBase   = nq * 256;
    const int m    = lane & 15;       // A row / B&C column within the 16x16 tile
    const int g    = lane >> 4;       // lane group

    float8 c[16];

    // ================= layer 0: [64x512] @ [512x1024] =================
    #pragma unroll
    for (int t = 0; t < 16; ++t) {
        #pragma unroll
        for (int i = 0; i < 8; ++i) c[t][i] = 0.f;
    }
    for (int k0 = 0; k0 < INF; k0 += 32) {
        half16 a;
        {
            half8 alo = *(const half8*)&ldsIn[(rowBase + m) * IN_STRIDE + k0 + 8 * g];
            half8 ahi = *(const half8*)&ldsIn[(rowBase + m) * IN_STRIDE + k0 + 16 + 8 * g];
            #pragma unroll
            for (int i = 0; i < 8; ++i) { a[i] = alo[i]; a[8 + i] = ahi[i]; }
        }
        #pragma unroll
        for (int t = 0; t < 16; ++t) {
            int col = nBase + t * 16 + m;
            const _Float16* bp = W0 + (size_t)col * INF + k0 + 8 * g;
            half8 blo = *(const half8*)bp;
            half8 bhi = *(const half8*)(bp + 16);
            half16 b;
            #pragma unroll
            for (int i = 0; i < 8; ++i) { b[i] = blo[i]; b[8 + i] = bhi[i]; }
            c[t] = __builtin_amdgcn_wmma_f32_16x16x32_f16(
                       false, a, false, b, (short)0, c[t], false, false);
        }
    }
    // epilogue: +bias, relu, f16 -> LDS h1
    #pragma unroll
    for (int t = 0; t < 16; ++t) {
        int col = nBase + t * 16 + m;
        float bias = B0[col];
        #pragma unroll
        for (int r = 0; r < 8; ++r) {
            int row = rowBase + r + 8 * g;
            float v = c[t][r] + bias;
            v = v > 0.f ? v : 0.f;
            ldsH1[row * H1_STRIDE + col] = (_Float16)v;
        }
    }
    __syncthreads();

    // ================= layer 1: [64x1024] @ [1024x1024] =================
    #pragma unroll
    for (int t = 0; t < 16; ++t) {
        #pragma unroll
        for (int i = 0; i < 8; ++i) c[t][i] = 0.f;
    }
    for (int k0 = 0; k0 < WW; k0 += 32) {
        half16 a;
        {
            half8 alo = *(const half8*)&ldsH1[(rowBase + m) * H1_STRIDE + k0 + 8 * g];
            half8 ahi = *(const half8*)&ldsH1[(rowBase + m) * H1_STRIDE + k0 + 16 + 8 * g];
            #pragma unroll
            for (int i = 0; i < 8; ++i) { a[i] = alo[i]; a[8 + i] = ahi[i]; }
        }
        #pragma unroll
        for (int t = 0; t < 16; ++t) {
            int col = nBase + t * 16 + m;
            const _Float16* bp = W1 + (size_t)col * WW + k0 + 8 * g;
            half8 blo = *(const half8*)bp;
            half8 bhi = *(const half8*)(bp + 16);
            half16 b;
            #pragma unroll
            for (int i = 0; i < 8; ++i) { b[i] = blo[i]; b[8 + i] = bhi[i]; }
            c[t] = __builtin_amdgcn_wmma_f32_16x16x32_f16(
                       false, a, false, b, (short)0, c[t], false, false);
        }
    }
    // epilogue: +bias, relu, gate-weighted atomic accumulate
    #pragma unroll
    for (int t = 0; t < 16; ++t) {
        int col = nBase + t * 16 + m;
        float bias = B1[col];
        #pragma unroll
        for (int r = 0; r < 8; ++r) {
            int row = rowBase + r + 8 * g;
            float gw = ldsGw[row];
            if (gw != 0.f) {
                float v = c[t][r] + bias;
                v = v > 0.f ? v : 0.f;
                atomicAdd(&acc[(size_t)(bBase + row) * WW + col], gw * v);
            }
        }
    }
}

// ---------------- head: out[b] = tanh(acc[b]) @ hk[task_b] + hb[task_b] ----------------
__global__ __launch_bounds__(128) void head_kernel(
    const float* __restrict__ acc, const float* __restrict__ hk,
    const float* __restrict__ hb, const int* __restrict__ taskIdx,
    float* __restrict__ out)
{
    __shared__ float th[WW];
    __shared__ float partial[HD];
    int b   = blockIdx.x;
    int tid = threadIdx.x;
    for (int i = tid; i < WW; i += 128) th[i] = tanhf(acc[(size_t)b * WW + i]);
    __syncthreads();
    int t    = taskIdx[b];
    int d    = tid & 63;
    int part = tid >> 6;
    const float* hkt = hk + (size_t)t * WW * HD;
    float s = 0.f;
    for (int n = part * 512; n < part * 512 + 512; ++n)
        s += th[n] * hkt[(size_t)n * HD + d];
    if (part == 1) partial[d] = s;
    __syncthreads();
    if (part == 0) out[(size_t)b * HD + d] = s + partial[d] + hb[t * HD + d];
}

// ---------------- host launcher ----------------
extern "C" void kernel_launch(void* const* d_in, const int* in_sizes, int n_in,
                              void* d_out, int out_size, void* d_ws, size_t ws_size,
                              hipStream_t stream) {
    const float* x   = (const float*)d_in[0];
    const float* gk  = (const float*)d_in[1];
    const float* rk0 = (const float*)d_in[2];
    const float* rb0 = (const float*)d_in[3];
    const float* rk1 = (const float*)d_in[4];
    const float* rb1 = (const float*)d_in[5];
    const float* sk0 = (const float*)d_in[6];
    const float* sb0 = (const float*)d_in[7];
    const float* sk1 = (const float*)d_in[8];
    const float* sb1 = (const float*)d_in[9];
    const float* hk  = (const float*)d_in[10];
    const float* hb  = (const float*)d_in[11];
    float* out = (float*)d_out;

    // workspace carve-up (256B aligned)
    char* ws = (char*)d_ws;
    size_t off = 0;
    auto take = [&](size_t bytes) -> char* {
        char* p = ws + off;
        off += (bytes + 255) & ~(size_t)255;
        return p;
    };
    _Float16* featH = (_Float16*)take((size_t)BB * INF * 2);        //  4.2 MB
    _Float16* r0T   = (_Float16*)take((size_t)EE * WW * INF * 2);   // 16.8 MB
    _Float16* r1T   = (_Float16*)take((size_t)EE * WW * WW * 2);    // 33.6 MB
    _Float16* s0T   = (_Float16*)take((size_t)SS * WW * INF * 2);   //  2.1 MB
    _Float16* s1T   = (_Float16*)take((size_t)SS * WW * WW * 2);    //  4.2 MB
    float*    gateW = (float*)take((size_t)BB * EE * 4);            //  0.3 MB
    int*      tIdx  = (int*)take((size_t)BB * 4);                   //  16 KB
    float*    acc   = (float*)take((size_t)BB * WW * 4);            // 16.8 MB

    // zero the accumulation buffer
    (void)hipMemsetAsync(acc, 0, (size_t)BB * WW * 4, stream);

    // gating + conversions
    gating_kernel<<<BB / 256, 256, 0, stream>>>(x, gk, gateW, tIdx);
    cvt_feat_kernel<<<(BB * INF) / 256, 256, 0, stream>>>(x, featH);
    tcvt_kernel<<<(EE * INF * WW) / 256, 256, 0, stream>>>(rk0, r0T, INF, WW);
    tcvt_kernel<<<(EE * WW  * WW) / 256, 256, 0, stream>>>(rk1, r1T, WW,  WW);
    tcvt_kernel<<<(SS * INF * WW) / 256, 256, 0, stream>>>(sk0, s0T, INF, WW);
    tcvt_kernel<<<(SS * WW  * WW) / 256, 256, 0, stream>>>(sk1, s1T, WW,  WW);

    // fused expert MLPs (16 routed + 2 shared)
    dim3 grid(BB / TM, EE + SS);
    expert_kernel<<<grid, NTHREADS, 0, stream>>>(featH, r0T, rb0, r1T, rb1,
                                                 s0T, sb0, s1T, sb1, gateW, acc);

    // tanh + per-task head
    head_kernel<<<BB, 128, 0, stream>>>(acc, hk, hb, tIdx, out);
}
